// MultiHeadedAttention_12678743457908
// MI455X (gfx1250) — matrix operand order repaired and checked by reading, not compiled, not run on hardware
//
#include <hip/hip_runtime.h>
#include <hip/hip_bf16.h>

// ---------------- problem constants ----------------
#define BATCH   4
#define SLEN    2048
#define DMODEL  1024
#define NHEADS  16
#define DHEAD   64
#define MROWS   (BATCH * SLEN)   // 8192

typedef __bf16 bf16x16 __attribute__((ext_vector_type(16)));
typedef __bf16 bf16x8  __attribute__((ext_vector_type(8)));
typedef float  v8f     __attribute__((ext_vector_type(8)));
typedef int    i32x4   __attribute__((vector_size(16)));

union BF2 { unsigned int u; __bf16 h[2]; };

#ifndef __has_builtin
#define __has_builtin(x) 0
#endif
#if __has_builtin(__builtin_amdgcn_global_load_async_to_lds_b128) && \
    __has_builtin(__builtin_amdgcn_s_wait_asynccnt)
#define HAVE_ASYNC_LDS 1
#else
#define HAVE_ASYNC_LDS 0
#endif

#define WMMA_BF16(a, b, c) \
    __builtin_amdgcn_wmma_f32_16x16x32_bf16(false, (a), false, (b), (short)0, (c), false, false)

__device__ inline unsigned int pack2(float x, float y) {
    BF2 t; t.h[0] = (__bf16)x; t.h[1] = (__bf16)y; return t.u;
}
__device__ inline bf16x16 cat8(bf16x8 lo, bf16x8 hi) {
    return __builtin_shufflevector(lo, hi, 0,1,2,3,4,5,6,7,8,9,10,11,12,13,14,15);
}

#if HAVE_ASYNC_LDS
// global_load_async_to_lds_b128: (v4i AS1* src, v4i AS3* dst, imm offset, imm cpol)
__device__ inline void async_copy_b128(const void* g, void* l) {
    __builtin_amdgcn_global_load_async_to_lds_b128(
        (__attribute__((address_space(1))) i32x4*)(uintptr_t)g,
        (__attribute__((address_space(3))) i32x4*)(unsigned int)(uintptr_t)l,
        0, 0);
}
#endif

// =====================================================================
// Kernel 1: fused Q/K/V projection GEMM.
//   X[8192,1024] fp32 x W[1024,1024] fp32 -> bf16 [B][H][S][DHEAD].
// grid=(M/64, N/128, 3), block=256 (8 waves); each wave: 32x32 tile.
// LDS: A natural [m][k]; B transposed [n][k] so fragments are contiguous.
// =====================================================================
__global__ __launch_bounds__(256) void mha_qkv_gemm(
    const float* __restrict__ q, const float* __restrict__ k, const float* __restrict__ v,
    const float* __restrict__ Wq, const float* __restrict__ bq,
    const float* __restrict__ Wk, const float* __restrict__ bk,
    const float* __restrict__ Wv, const float* __restrict__ bv,
    __bf16* __restrict__ qo, __bf16* __restrict__ ko, __bf16* __restrict__ vo)
{
    const int z = blockIdx.z;
    const float* X    = (z == 0) ? q  : (z == 1) ? k  : v;
    const float* W    = (z == 0) ? Wq : (z == 1) ? Wk : Wv;
    const float* bias = (z == 0) ? bq : (z == 1) ? bk : bv;
    __bf16*      out  = (z == 0) ? qo : (z == 1) ? ko : vo;

    __shared__ __bf16 sA [64][40];    // [m][k]
    __shared__ __bf16 sBT[128][40];   // [n][k] (transposed)

    const int lane  = threadIdx.x & 31;
    const int wave  = threadIdx.x >> 5;
    const int col16 = lane & 15;
    const int half  = lane >> 4;
    const int mb    = (wave >> 2) * 32;   // 0 / 32
    const int nb    = (wave &  3) * 32;   // 0 / 32 / 64 / 96

    const int m0 = blockIdx.x * 64;
    const int n0 = blockIdx.y * 128;

    v8f c00 = {}, c01 = {}, c10 = {}, c11 = {};

    for (int kb = 0; kb < DMODEL; kb += 32) {
        __syncthreads();
        {
            const int t = threadIdx.x;
            // ---- A tile 64x32 fp32 -> bf16, one b128 LDS store/thread ----
            int r = t >> 2, c = (t & 3) * 8;
            const float* ap = X + (size_t)(m0 + r) * DMODEL + kb + c;
            float4 f0 = *(const float4*)ap;
            float4 f1 = *(const float4*)(ap + 4);
            uint4 pk;
            pk.x = pack2(f0.x, f0.y); pk.y = pack2(f0.z, f0.w);
            pk.z = pack2(f1.x, f1.y); pk.w = pack2(f1.z, f1.w);
            *(uint4*)&sA[r][c] = pk;
            // ---- B tile 32x128 fp32 -> bf16 transposed [n][k] ----
            int kp = t >> 4;            // k pair: k = 2*kp
            int nc = (t & 15) * 8;      // 8 n-columns
            const float* bp0 = W + (size_t)(kb + 2 * kp) * DMODEL + n0 + nc;
            const float* bp1 = bp0 + DMODEL;
            float4 g0 = *(const float4*)bp0;  float4 g1 = *(const float4*)(bp0 + 4);
            float4 h0 = *(const float4*)bp1;  float4 h1 = *(const float4*)(bp1 + 4);
            *(unsigned int*)&sBT[nc + 0][2 * kp] = pack2(g0.x, h0.x);
            *(unsigned int*)&sBT[nc + 1][2 * kp] = pack2(g0.y, h0.y);
            *(unsigned int*)&sBT[nc + 2][2 * kp] = pack2(g0.z, h0.z);
            *(unsigned int*)&sBT[nc + 3][2 * kp] = pack2(g0.w, h0.w);
            *(unsigned int*)&sBT[nc + 4][2 * kp] = pack2(g1.x, h1.x);
            *(unsigned int*)&sBT[nc + 5][2 * kp] = pack2(g1.y, h1.y);
            *(unsigned int*)&sBT[nc + 6][2 * kp] = pack2(g1.z, h1.z);
            *(unsigned int*)&sBT[nc + 7][2 * kp] = pack2(g1.w, h1.w);
            if (kb + 32 < DMODEL) {                     // prefetch next K-step
                __builtin_prefetch(ap + 32, 0, 3);
                __builtin_prefetch(bp0 + (size_t)32 * DMODEL, 0, 3);
            }
        }
        __syncthreads();

        // ---- fragments: all contiguous vector LDS loads ----
        const int row0 = mb + col16, row1 = mb + 16 + col16;
        bf16x16 a0 = cat8(*(const bf16x8*)&sA[row0][16 * half],
                          *(const bf16x8*)&sA[row0][16 * half + 8]);
        bf16x16 a1 = cat8(*(const bf16x8*)&sA[row1][16 * half],
                          *(const bf16x8*)&sA[row1][16 * half + 8]);
        bf16x16 b0 = *(const bf16x16*)&sBT[nb + col16][16 * half];
        bf16x16 b1 = *(const bf16x16*)&sBT[nb + 16 + col16][16 * half];

        c00 = WMMA_BF16(a0, b0, c00);
        c01 = WMMA_BF16(a0, b1, c01);
        c10 = WMMA_BF16(a1, b0, c10);
        c11 = WMMA_BF16(a1, b1, c11);
    }

    // ---- epilogue: bias add, scatter bf16 into [B][H][S][DHEAD] ----
    #pragma unroll
    for (int mi = 0; mi < 2; ++mi) {
        #pragma unroll
        for (int ni = 0; ni < 2; ++ni) {
            v8f cc = mi ? (ni ? c11 : c10) : (ni ? c01 : c00);
            const int ncol = n0 + nb + ni * 16 + col16;
            const float bval = bias[ncol];
            const int hh = ncol >> 6, dd = ncol & 63;
            #pragma unroll
            for (int r = 0; r < 8; ++r) {
                int m  = m0 + mb + mi * 16 + r + 8 * half;
                int bi = m >> 11;
                int ss = m & (SLEN - 1);
                out[(((size_t)bi * NHEADS + hh) * SLEN + ss) * DHEAD + dd] =
                    (__bf16)(cc[r] + bval);
            }
        }
    }
}

// =====================================================================
// Kernel 2: flash attention (online softmax), bf16 WMMA.
// grid=(S/64, B*H), block=128 (4 waves); each wave owns 16 q rows.
// K stored [t][d] (logits B-frags contiguous); V stored transposed
// [d][t] (PV B-frags contiguous).
// =====================================================================
__global__ __launch_bounds__(128) void mha_flash_attn(
    const __bf16* __restrict__ Q, const __bf16* __restrict__ K,
    const __bf16* __restrict__ V, const unsigned char* __restrict__ mask,
    __bf16* __restrict__ Oa)
{
    constexpr int Tc = 32;

    const int lane  = threadIdx.x & 31;
    const int wave  = threadIdx.x >> 5;
    const int col16 = lane & 15;
    const int half  = lane >> 4;

    const int bh = blockIdx.y;
    const int b  = bh >> 4;
    const int h  = bh & (NHEADS - 1);
    const int q0 = blockIdx.x * 64 + wave * 16;

    __shared__ __bf16 sK [Tc][DHEAD + 8];   // [t][d]
    __shared__ __bf16 sVT[DHEAD][Tc + 8];   // [d][t]
    __shared__ __bf16 sP [4][16][Tc + 8];   // per-wave P scratch

    const size_t base = (size_t)bh * SLEN * DHEAD;

    // ---- resident Q A-fragments (16x64) straight from global, b128s ----
    bf16x16 aq0, aq1;
    {
        const __bf16* qrow = Q + base + (size_t)(q0 + col16) * DHEAD;
        aq0 = cat8(*(const bf16x8*)(qrow + 16 * half),
                   *(const bf16x8*)(qrow + 8 + 16 * half));
        aq1 = cat8(*(const bf16x8*)(qrow + 32 + 16 * half),
                   *(const bf16x8*)(qrow + 40 + 16 * half));
    }

    v8f o0 = {}, o1 = {}, o2 = {}, o3 = {};
    float m_i[8], l_i[8];
    #pragma unroll
    for (int r = 0; r < 8; ++r) { m_i[r] = -1e30f; l_i[r] = 0.0f; }

    const unsigned char* mbase = mask + (size_t)b * SLEN * SLEN;

    for (int t0 = 0; t0 < SLEN; t0 += Tc) {
        __syncthreads();
        {
            const int t = threadIdx.x;
            // ---- K tile 32x64: async DMA to LDS when available ----
            int r = t >> 2, c = (t & 3) * 16;
            const __bf16* kg = K + base + (size_t)(t0 + r) * DHEAD + c;
#if HAVE_ASYNC_LDS
            async_copy_b128(kg,     &sK[r][c]);
            async_copy_b128(kg + 8, &sK[r][c + 8]);
#else
            *(uint4*)&sK[r][c]     = *(const uint4*)kg;
            *(uint4*)&sK[r][c + 8] = *(const uint4*)(kg + 8);
#endif
            // ---- V tile 32x64 -> transposed [d][t], packed pair stores ----
            int tt = (t >> 3) * 2, d0 = (t & 7) * 8;
            const __bf16* vg = V + base + (size_t)(t0 + tt) * DHEAD + d0;
            bf16x8 r0 = *(const bf16x8*)vg;
            bf16x8 r1 = *(const bf16x8*)(vg + DHEAD);
            #pragma unroll
            for (int j = 0; j < 8; ++j) {
                BF2 p; p.h[0] = r0[j]; p.h[1] = r1[j];
                *(unsigned int*)&sVT[d0 + j][tt] = p.u;
            }
            if (t0 + Tc < SLEN) {                      // prefetch next tile
                __builtin_prefetch(kg + (size_t)Tc * DHEAD, 0, 3);
                __builtin_prefetch(vg + (size_t)Tc * DHEAD, 0, 3);
            }
        }
#if HAVE_ASYNC_LDS
        __builtin_amdgcn_s_wait_asynccnt(0);
#endif
        __syncthreads();

        // ---- logits S = Q K^T : 16x32 ----
        v8f s0 = {}, s1 = {};
        #pragma unroll
        for (int ks = 0; ks < 2; ++ks) {
            bf16x16 b0 = *(const bf16x16*)&sK[col16][16 * half + 32 * ks];
            bf16x16 b1 = *(const bf16x16*)&sK[16 + col16][16 * half + 32 * ks];
            const bf16x16 aq = ks ? aq1 : aq0;
            s0 = WMMA_BF16(aq, b0, s0);
            s1 = WMMA_BF16(aq, b1, s1);
        }

        // ---- mask: where(mask) -> -1e8 ----
        #pragma unroll
        for (int r = 0; r < 8; ++r) {
            int qr = q0 + r + 8 * half;
            const unsigned char* mp = mbase + (size_t)qr * SLEN + t0;
            if (mp[col16])      s0[r] = -1e8f;
            if (mp[16 + col16]) s1[r] = -1e8f;
        }

        // ---- online softmax (16-lane half reductions) ----
        #pragma unroll
        for (int r = 0; r < 8; ++r) {
            float mx = fmaxf(s0[r], s1[r]);
            #pragma unroll
            for (int off = 8; off >= 1; off >>= 1)
                mx = fmaxf(mx, __shfl_xor(mx, off, 32));
            float mnew = fmaxf(m_i[r], mx);
            float corr = __expf(m_i[r] - mnew);
            float p0 = __expf(s0[r] - mnew);
            float p1 = __expf(s1[r] - mnew);
            s0[r] = p0; s1[r] = p1;
            float rs = p0 + p1;
            #pragma unroll
            for (int off = 8; off >= 1; off >>= 1)
                rs += __shfl_xor(rs, off, 32);
            l_i[r] = l_i[r] * corr + rs;
            m_i[r] = mnew;
            o0[r] *= corr; o1[r] *= corr; o2[r] *= corr; o3[r] *= corr;
        }

        // ---- P: C-layout -> A-layout via wave-private LDS ----
        #pragma unroll
        for (int r = 0; r < 8; ++r) {
            sP[wave][r + 8 * half][col16]      = (__bf16)s0[r];
            sP[wave][r + 8 * half][16 + col16] = (__bf16)s1[r];
        }
        bf16x16 ap = cat8(*(const bf16x8*)&sP[wave][col16][16 * half],
                          *(const bf16x8*)&sP[wave][col16][16 * half + 8]);

        // ---- O += P (16x32) x V (32x64) ----
        bf16x16 bv0 = *(const bf16x16*)&sVT[col16][16 * half];
        bf16x16 bv1 = *(const bf16x16*)&sVT[16 + col16][16 * half];
        bf16x16 bv2 = *(const bf16x16*)&sVT[32 + col16][16 * half];
        bf16x16 bv3 = *(const bf16x16*)&sVT[48 + col16][16 * half];
        o0 = WMMA_BF16(ap, bv0, o0);
        o1 = WMMA_BF16(ap, bv1, o1);
        o2 = WMMA_BF16(ap, bv2, o2);
        o3 = WMMA_BF16(ap, bv3, o3);
    }

    // ---- epilogue: /l, * 1/sqrt(64); store [B][S][H*DHEAD] bf16 ----
    #pragma unroll
    for (int r = 0; r < 8; ++r) {
        float inv = 0.125f / l_i[r];
        int s = q0 + r + 8 * half;
        __bf16* op = Oa + ((size_t)b * SLEN + s) * DMODEL + h * DHEAD;
        op[col16]      = (__bf16)(o0[r] * inv);
        op[16 + col16] = (__bf16)(o1[r] * inv);
        op[32 + col16] = (__bf16)(o2[r] * inv);
        op[48 + col16] = (__bf16)(o3[r] * inv);
    }
}

// =====================================================================
// Kernel 3: output projection. attn[8192,1024] bf16 x Wo[1024,1024] fp32
//   -> out fp32 + bo. Same tiling as kernel 1.
// =====================================================================
__global__ __launch_bounds__(256) void mha_out_gemm(
    const __bf16* __restrict__ A, const float* __restrict__ Wo,
    const float* __restrict__ bo, float* __restrict__ out)
{
    __shared__ __bf16 sA [64][40];
    __shared__ __bf16 sBT[128][40];

    const int lane  = threadIdx.x & 31;
    const int wave  = threadIdx.x >> 5;
    const int col16 = lane & 15;
    const int half  = lane >> 4;
    const int mb    = (wave >> 2) * 32;
    const int nb    = (wave &  3) * 32;

    const int m0 = blockIdx.x * 64;
    const int n0 = blockIdx.y * 128;

    v8f c00 = {}, c01 = {}, c10 = {}, c11 = {};

    for (int kb = 0; kb < DMODEL; kb += 32) {
        __syncthreads();
        {
            const int t = threadIdx.x;
            // A tile already bf16: one b128 copy
            int r = t >> 2, c = (t & 3) * 8;
            const __bf16* ap = A + (size_t)(m0 + r) * DMODEL + kb + c;
            *(uint4*)&sA[r][c] = *(const uint4*)ap;
            // B tile fp32 -> bf16 transposed
            int kp = t >> 4, nc = (t & 15) * 8;
            const float* bp0 = Wo + (size_t)(kb + 2 * kp) * DMODEL + n0 + nc;
            const float* bp1 = bp0 + DMODEL;
            float4 g0 = *(const float4*)bp0;  float4 g1 = *(const float4*)(bp0 + 4);
            float4 h0 = *(const float4*)bp1;  float4 h1 = *(const float4*)(bp1 + 4);
            *(unsigned int*)&sBT[nc + 0][2 * kp] = pack2(g0.x, h0.x);
            *(unsigned int*)&sBT[nc + 1][2 * kp] = pack2(g0.y, h0.y);
            *(unsigned int*)&sBT[nc + 2][2 * kp] = pack2(g0.z, h0.z);
            *(unsigned int*)&sBT[nc + 3][2 * kp] = pack2(g0.w, h0.w);
            *(unsigned int*)&sBT[nc + 4][2 * kp] = pack2(g1.x, h1.x);
            *(unsigned int*)&sBT[nc + 5][2 * kp] = pack2(g1.y, h1.y);
            *(unsigned int*)&sBT[nc + 6][2 * kp] = pack2(g1.z, h1.z);
            *(unsigned int*)&sBT[nc + 7][2 * kp] = pack2(g1.w, h1.w);
            if (kb + 32 < DMODEL) {
                __builtin_prefetch(ap + 32, 0, 3);
                __builtin_prefetch(bp0 + (size_t)32 * DMODEL, 0, 3);
            }
        }
        __syncthreads();

        const int row0 = mb + col16, row1 = mb + 16 + col16;
        bf16x16 a0 = cat8(*(const bf16x8*)&sA[row0][16 * half],
                          *(const bf16x8*)&sA[row0][16 * half + 8]);
        bf16x16 a1 = cat8(*(const bf16x8*)&sA[row1][16 * half],
                          *(const bf16x8*)&sA[row1][16 * half + 8]);
        bf16x16 b0 = *(const bf16x16*)&sBT[nb + col16][16 * half];
        bf16x16 b1 = *(const bf16x16*)&sBT[nb + 16 + col16][16 * half];

        c00 = WMMA_BF16(a0, b0, c00);
        c01 = WMMA_BF16(a0, b1, c01);
        c10 = WMMA_BF16(a1, b0, c10);
        c11 = WMMA_BF16(a1, b1, c11);
    }

    #pragma unroll
    for (int mi = 0; mi < 2; ++mi) {
        #pragma unroll
        for (int ni = 0; ni < 2; ++ni) {
            v8f cc = mi ? (ni ? c11 : c10) : (ni ? c01 : c00);
            const int ncol = n0 + nb + ni * 16 + col16;
            const float bval = bo[ncol];
            #pragma unroll
            for (int r = 0; r < 8; ++r) {
                int m = m0 + mb + mi * 16 + r + 8 * half;
                out[(size_t)m * DMODEL + ncol] = cc[r] + bval;
            }
        }
    }
}

// =====================================================================
extern "C" void kernel_launch(void* const* d_in, const int* in_sizes, int n_in,
                              void* d_out, int out_size, void* d_ws, size_t ws_size,
                              hipStream_t stream)
{
    const float* query = (const float*)d_in[0];
    const float* key_  = (const float*)d_in[1];
    const float* value = (const float*)d_in[2];
    const unsigned char* mask = (const unsigned char*)d_in[3];  // jnp bool -> 1 byte
    const float* Wq = (const float*)d_in[4];
    const float* bq = (const float*)d_in[5];
    const float* Wk = (const float*)d_in[6];
    const float* bk = (const float*)d_in[7];
    const float* Wv = (const float*)d_in[8];
    const float* bv = (const float*)d_in[9];
    const float* Wo = (const float*)d_in[10];
    const float* bo = (const float*)d_in[11];
    float* out = (float*)d_out;

    char* ws = (char*)d_ws;
    const size_t SEG = (size_t)MROWS * DMODEL * sizeof(__bf16);  // 16 MB
    __bf16* qB   = (__bf16*)(ws);
    __bf16* kB   = (__bf16*)(ws + SEG);
    __bf16* vB   = (__bf16*)(ws + 2 * SEG);
    __bf16* attn = (__bf16*)(ws + 3 * SEG);

    dim3 g1(MROWS / 64, DMODEL / 128, 3);
    mha_qkv_gemm<<<g1, 256, 0, stream>>>(query, key_, value,
                                         Wq, bq, Wk, bk, Wv, bv, qB, kB, vB);

    dim3 g2(SLEN / 64, BATCH * NHEADS);
    mha_flash_attn<<<g2, 128, 0, stream>>>(qB, kB, vB, mask, attn);

    dim3 g3(MROWS / 64, DMODEL / 128);
    mha_out_gemm<<<g3, 256, 0, stream>>>(attn, Wo, bo, out);
}
